// DIN_35914516529539
// MI455X (gfx1250) — matrix operand order repaired and checked by reading, not compiled, not run on hardware
//
#include <hip/hip_runtime.h>
#include <hip/hip_bf16.h>
#include <math.h>

typedef __attribute__((ext_vector_type(16))) _Float16 v16h;
typedef __attribute__((ext_vector_type(8)))  float    v8f;

#define B_SAMPLES 16384
#define DDIM      16
#define WAVES     8       // waves per 256-thread block

// Intra-wave LDS handoff: DS ops are in-order per wave and lanes are lockstep,
// so a wavefront-scope fence (compiler ordering) + wave_barrier (scheduling
// barrier) is sufficient -- no workgroup barrier, allowing per-wave early exit.
__device__ __forceinline__ void wave_sync() {
    __builtin_amdgcn_fence(__ATOMIC_RELEASE, "wavefront");
    __builtin_amdgcn_wave_barrier();
    __builtin_amdgcn_fence(__ATOMIC_ACQUIRE, "wavefront");
}

// ---------------------------------------------------------------------------
// Attention kernel: one wave owns one sample; loops over ceil(len/16) tiles.
// Weights (aw0 -> B0/B1, aw1 -> B2) are converted to f16 fragments ONCE and
// held in registers across the tile loop.
//   layer0: [q,k,q-k,q*k](64) -> 16 : 2x v_wmma_f32_16x16x32_f16 per tile
//   layer1: 16 -> 8 (zero padded)   : 1x v_wmma_f32_16x16x32_f16 per tile
//   layer2: 8 -> 1                  : scalar
//   pooled: register accumulation across tiles, shfl reduce, single store
// ---------------------------------------------------------------------------
__global__ __launch_bounds__(256) void din_attn(
    const float* __restrict__ x,
    const int*   __restrict__ seq_off,
    const int*   __restrict__ seq_len,
    const int*   __restrict__ target_off,
    const float* __restrict__ aw0, const float* __restrict__ ab0,
    const float* __restrict__ aw1, const float* __restrict__ ab1,
    const float* __restrict__ aw2, const float* __restrict__ ab2,
    float* __restrict__ pooled)
{
    __shared__ float lds_h[WAVES][16][20];   // 80B rows: 16B aligned, conflict-free
    __shared__ float lds_s[WAVES][16];       // per-position scores

    const int wave = threadIdx.x >> 5;
    const int lane = threadIdx.x & 31;
    const int m  = lane & 15;                // row (position) / col (unit)
    const int kh = lane >> 4;                // K-half select
    const int b  = blockIdx.x * WAVES + wave;   // exactly B_SAMPLES waves

    const int len  = seq_len[b];
    const int soff = seq_off[b];

    // --- loop-invariant loads: q slice + all weight fragments (f16, once) ---
    const float* qp = x + (size_t)target_off[b] * DDIM + kh * 8;
    float q[8];
    #pragma unroll
    for (int j = 0; j < 8; ++j) q[j] = qp[j];

    v16h B0, B1, B2;                         // aw0 chunk0, aw0 chunk1, aw1 (padded)
    #pragma unroll
    for (int j = 0; j < 16; ++j) {
        const int K = kh * 8 + (j & 7) + (j >> 3) * 16;
        B0[j] = (_Float16)aw0[K * 16 + m];
        B1[j] = (_Float16)aw0[(32 + K) * 16 + m];
        B2[j] = (K < 16 && m < 8) ? (_Float16)aw1[K * 8 + m] : (_Float16)0.0f;
    }
    const float bias0 = ab0[m];
    const float bias1 = (m < 8) ? ab1[m] : 0.0f;
    float aw2r[8];
    #pragma unroll
    for (int u = 0; u < 8; ++u) aw2r[u] = aw2[u];   // uniform -> scalar loads
    const float ab2v = ab2[0];

    float acc_p[8];                          // pooled partials (this lane's 8 dims)
    #pragma unroll
    for (int j = 0; j < 8; ++j) acc_p[j] = 0.0f;

    const int ntile = (len + 15) >> 4;       // early exit past valid history
    for (int t = 0; t < ntile; ++t) {
        const int pos = t * 16 + m;
        float k8[8];
        if (pos < len) {
            const float* kp = x + (size_t)(soff + pos) * DDIM + kh * 8;
            #pragma unroll
            for (int j = 0; j < 8; ++j) k8[j] = kp[j];
        } else {
            #pragma unroll
            for (int j = 0; j < 8; ++j) k8[j] = 0.0f;
        }

        // A fragments: chunk0 = [q | k], chunk1 = [q-k | q*k]
        v16h A0, A1;
        #pragma unroll
        for (int j = 0; j < 8; ++j) {
            A0[j]     = (_Float16)q[j];
            A0[j + 8] = (_Float16)k8[j];
            A1[j]     = (_Float16)(q[j] - k8[j]);
            A1[j + 8] = (_Float16)(q[j] * k8[j]);
        }

        v8f acc;
        #pragma unroll
        for (int r = 0; r < 8; ++r) acc[r] = bias0;
        acc = __builtin_amdgcn_wmma_f32_16x16x32_f16(false, A0, false, B0, (short)0, acc, false, false);
        acc = __builtin_amdgcn_wmma_f32_16x16x32_f16(false, A1, false, B1, (short)0, acc, false, false);
        #pragma unroll
        for (int r = 0; r < 8; ++r) lds_h[wave][r + 8 * kh][m] = fmaxf(acc[r], 0.0f);
        wave_sync();

        // layer1: A = h0 (K=16 valid, upper half zero)
        v16h A2;
        #pragma unroll
        for (int j = 0; j < 8; ++j) {
            A2[j]     = (_Float16)lds_h[wave][m][kh * 8 + j];
            A2[j + 8] = (_Float16)0.0f;
        }
        v8f acc2;
        #pragma unroll
        for (int r = 0; r < 8; ++r) acc2[r] = bias1;
        acc2 = __builtin_amdgcn_wmma_f32_16x16x32_f16(false, A2, false, B2, (short)0, acc2, false, false);
        wave_sync();                          // all A2 reads done before overwrite
        #pragma unroll
        for (int r = 0; r < 8; ++r) lds_h[wave][r + 8 * kh][m] = fmaxf(acc2[r], 0.0f);
        wave_sync();

        // layer2 scores (lanes 0..15, one position each)
        if (kh == 0) {
            float s = ab2v;
            #pragma unroll
            for (int u = 0; u < 8; ++u) s += lds_h[wave][m][u] * aw2r[u];
            lds_s[wave][m] = s;
        }
        wave_sync();

        const float sc = lds_s[wave][m];
        #pragma unroll
        for (int j = 0; j < 8; ++j) acc_p[j] += sc * k8[j];  // invalid keys are 0
    }

    // reduce pooled over the 16 positions-lanes of each K-half group
    #pragma unroll
    for (int off = 1; off < 16; off <<= 1) {
        #pragma unroll
        for (int j = 0; j < 8; ++j) acc_p[j] += __shfl_xor(acc_p[j], off, 32);
    }
    if (m == 0) {
        float4* dst = (float4*)(pooled + (size_t)b * DDIM + kh * 8);
        dst[0] = make_float4(acc_p[0], acc_p[1], acc_p[2], acc_p[3]);
        dst[1] = make_float4(acc_p[4], acc_p[5], acc_p[6], acc_p[7]);
    }
}

// ---------------------------------------------------------------------------
// Top MLP: one wave handles 16 samples. z = [other | pooled | target] (48)
//   layer0: 48 -> 32 : 2 N-tiles x 2 K-chunks of v_wmma_f32_16x16x32_f16
//   layer1: 32 -> 16 : 1x v_wmma_f32_16x16x32_f16
//   layer2: 16 -> 1  : scalar + sigmoid
// ---------------------------------------------------------------------------
__global__ __launch_bounds__(256) void din_top(
    const float* __restrict__ x,
    const int*   __restrict__ other_off,
    const int*   __restrict__ target_off,
    const float* __restrict__ pooled,
    const float* __restrict__ mw0, const float* __restrict__ mb0,
    const float* __restrict__ mw1, const float* __restrict__ mb1,
    const float* __restrict__ mw2, const float* __restrict__ mb2,
    float* __restrict__ out)
{
    __shared__ float lds_h[WAVES][16][36];   // 144B rows: 16B aligned, conflict-free

    const int wave = threadIdx.x >> 5;
    const int lane = threadIdx.x & 31;
    const int m  = lane & 15;
    const int kh = lane >> 4;
    const int s = (blockIdx.x * WAVES + wave) * 16 + m;   // sample index (row m)

    float oth[8], pld[8], tgt[8];
    const float* op = x + (size_t)other_off[s]  * DDIM + kh * 8;
    const float* tp = x + (size_t)target_off[s] * DDIM + kh * 8;
    const float* pp = pooled + (size_t)s * DDIM + kh * 8;
    #pragma unroll
    for (int j = 0; j < 8; ++j) { oth[j] = op[j]; tgt[j] = tp[j]; pld[j] = pp[j]; }

    // A chunk0 = [other | pooled] (K 0..31), chunk1 = [target | 0] (K 32..63)
    v16h A0, A1;
    #pragma unroll
    for (int j = 0; j < 8; ++j) {
        A0[j]     = (_Float16)oth[j];
        A0[j + 8] = (_Float16)pld[j];
        A1[j]     = (_Float16)tgt[j];
        A1[j + 8] = (_Float16)0.0f;          // K >= 48 padding
    }

    // layer0: units 0..31 in two N-tiles
    #pragma unroll
    for (int nt = 0; nt < 2; ++nt) {
        v16h B0, B1;
        #pragma unroll
        for (int j = 0; j < 16; ++j) {
            const int k0 = kh * 8 + (j & 7) + (j >> 3) * 16;       // feature for chunk0
            const int k1 = 32 + kh * 8 + (j & 7) + (j >> 3) * 16;  // feature for chunk1
            B0[j] = (_Float16)mw0[k0 * 32 + nt * 16 + m];
            B1[j] = (k1 < 48) ? (_Float16)mw0[k1 * 32 + nt * 16 + m] : (_Float16)0.0f;
        }
        v8f acc;
        const float bb = mb0[nt * 16 + m];
        #pragma unroll
        for (int r = 0; r < 8; ++r) acc[r] = bb;
        acc = __builtin_amdgcn_wmma_f32_16x16x32_f16(false, A0, false, B0, (short)0, acc, false, false);
        acc = __builtin_amdgcn_wmma_f32_16x16x32_f16(false, A1, false, B1, (short)0, acc, false, false);
        #pragma unroll
        for (int r = 0; r < 8; ++r) lds_h[wave][r + 8 * kh][nt * 16 + m] = fmaxf(acc[r], 0.0f);
    }
    __syncthreads();

    // layer1: A = h0 (16x32), B = mw1 (32x16)
    v16h A2, B2;
    #pragma unroll
    for (int j = 0; j < 8; ++j) {
        A2[j]     = (_Float16)lds_h[wave][m][kh * 8 + j];
        A2[j + 8] = (_Float16)lds_h[wave][m][16 + kh * 8 + j];
    }
    #pragma unroll
    for (int j = 0; j < 16; ++j) B2[j] = (_Float16)mw1[(kh * 8 + (j & 7) + (j >> 3) * 16) * 16 + m];
    v8f acc2;
    const float bb1 = mb1[m];
    #pragma unroll
    for (int r = 0; r < 8; ++r) acc2[r] = bb1;
    acc2 = __builtin_amdgcn_wmma_f32_16x16x32_f16(false, A2, false, B2, (short)0, acc2, false, false);
    __syncthreads();
    #pragma unroll
    for (int r = 0; r < 8; ++r) lds_h[wave][r + 8 * kh][m] = fmaxf(acc2[r], 0.0f);
    __syncthreads();

    // layer2 + sigmoid
    if (kh == 0) {
        float o = mb2[0];
        #pragma unroll
        for (int u = 0; u < 16; ++u) o += lds_h[wave][m][u] * mw2[u];
        out[s] = 1.0f / (1.0f + expf(-o));
    }
}

// ---------------------------------------------------------------------------
extern "C" void kernel_launch(void* const* d_in, const int* in_sizes, int n_in,
                              void* d_out, int out_size, void* d_ws, size_t ws_size,
                              hipStream_t stream) {
    (void)in_sizes; (void)n_in; (void)out_size; (void)ws_size;
    const float* x          = (const float*)d_in[0];
    const int*   other_off  = (const int*)  d_in[1];
    const int*   seq_off    = (const int*)  d_in[2];
    const int*   seq_len    = (const int*)  d_in[3];
    const int*   target_off = (const int*)  d_in[4];
    const float* aw0 = (const float*)d_in[5];
    const float* ab0 = (const float*)d_in[6];
    const float* aw1 = (const float*)d_in[7];
    const float* ab1 = (const float*)d_in[8];
    const float* aw2 = (const float*)d_in[9];
    const float* ab2 = (const float*)d_in[10];
    const float* mw0 = (const float*)d_in[11];
    const float* mb0 = (const float*)d_in[12];
    const float* mw1 = (const float*)d_in[13];
    const float* mb1 = (const float*)d_in[14];
    const float* mw2 = (const float*)d_in[15];
    const float* mb2 = (const float*)d_in[16];
    float* out    = (float*)d_out;
    float* pooled = (float*)d_ws;            // B*D floats, fully written (no init)

    din_attn<<<B_SAMPLES / WAVES, 256, 0, stream>>>(x, seq_off, seq_len, target_off,
                                                    aw0, ab0, aw1, ab1, aw2, ab2, pooled);

    din_top<<<(B_SAMPLES / 16) / WAVES, 256, 0, stream>>>(x, other_off, target_off, pooled,
                                                          mw0, mb0, mw1, mb1, mw2, mb2, out);
}
